// AttentionLayer_53635551593033
// MI455X (gfx1250) — matrix operand order repaired and checked by reading, not compiled, not run on hardware
//
#include <hip/hip_runtime.h>

// ---------------------------------------------------------------------------
// Types for CDNA5 WMMA (wave32): v_wmma_f32_16x16x32_bf16
// ---------------------------------------------------------------------------
typedef __attribute__((ext_vector_type(16))) __bf16 v16bf;
typedef __attribute__((ext_vector_type(8)))  __bf16 v8bf;
typedef __attribute__((ext_vector_type(2)))  __bf16 v2bf;
typedef __attribute__((ext_vector_type(8)))  float  v8f;
typedef int v4i_ __attribute__((vector_size(16)));   // b128 payload type

#define B_      16
#define T_      8192
#define D_      512
#define NH_     8
#define NKV_    2
#define HD_     64
#define NREP_   4
#define KVD_    128
#define CHUNK_T 128
#define NCHUNK  (T_ / CHUNK_T)   // 64
#define SCALE_  0.125f

// fp32 -> bf16, packed pair via v_cvt_pk_bf16_f32 when available.
static __device__ __forceinline__ __bf16 f2bf(float f) { return (__bf16)f; }

static __device__ __forceinline__ v2bf f2bf2(float x, float y) {
#if __has_builtin(__builtin_amdgcn_cvt_pk_bf16_f32)
  auto r = __builtin_amdgcn_cvt_pk_bf16_f32(x, y);
  return __builtin_bit_cast(v2bf, r);
#else
  v2bf p; p[0] = (__bf16)x; p[1] = (__bf16)y; return p;
#endif
}

static __device__ __forceinline__ v8f wmma_bf16(v16bf a, v16bf b, v8f c) {
  // 8 args: (neg_a, A, neg_b, B, c_mod, C, reuse_a, reuse_b)
  return __builtin_amdgcn_wmma_f32_16x16x32_bf16(false, a, false, b, (short)0, c,
                                                 false, false);
}

// ---------------------------------------------------------------------------
// Async global -> LDS copy of 16B per lane (GLOBAL_LOAD_ASYNC_TO_LDS_B128,
// tracked by ASYNCcnt).  The clang builtin takes generic v4i* pointers
// (per the round-3 diagnostic); falls back to a plain vector copy if absent.
// ---------------------------------------------------------------------------
#if __has_builtin(__builtin_amdgcn_global_load_async_to_lds_b128)
#define HAVE_ASYNC_LDS 1
#else
#define HAVE_ASYNC_LDS 0
#endif

static __device__ __forceinline__ void async_copy_b128(const void* gsrc,
                                                       void* ldst) {
#if HAVE_ASYNC_LDS
  __builtin_amdgcn_global_load_async_to_lds_b128((v4i_*)gsrc, (v4i_*)ldst, 0, 0);
#else
  *(v8bf*)ldst = *(const v8bf*)gsrc;
#endif
}

static __device__ __forceinline__ void wait_async_zero() {
#if HAVE_ASYNC_LDS
#if __has_builtin(__builtin_amdgcn_s_wait_asynccnt)
  __builtin_amdgcn_s_wait_asynccnt(0);
#else
  asm volatile("s_wait_asynccnt 0" ::: "memory");
#endif
#endif
}

// 16x32 bf16 A-tile slice from an fp32 row (contiguous layout:
// lane<16: K = 0..15 ; lane>=16: K = 16..31).
static __device__ __forceinline__ v16bf load_a_f32(const float* row, int kbase,
                                                   int khalf) {
  const float4* p0 = (const float4*)(row + kbase + khalf * 16);
  const float4* p1 = (const float4*)(row + kbase + khalf * 16 + 8);
  float4 x0 = p0[0], x1 = p0[1];
  float4 y0 = p1[0], y1 = p1[1];
  v16bf a;
  v2bf c;
  c = f2bf2(x0.x, x0.y); a[0]  = c[0]; a[1]  = c[1];
  c = f2bf2(x0.z, x0.w); a[2]  = c[0]; a[3]  = c[1];
  c = f2bf2(x1.x, x1.y); a[4]  = c[0]; a[5]  = c[1];
  c = f2bf2(x1.z, x1.w); a[6]  = c[0]; a[7]  = c[1];
  c = f2bf2(y0.x, y0.y); a[8]  = c[0]; a[9]  = c[1];
  c = f2bf2(y0.z, y0.w); a[10] = c[0]; a[11] = c[1];
  c = f2bf2(y1.x, y1.y); a[12] = c[0]; a[13] = c[1];
  c = f2bf2(y1.z, y1.w); a[14] = c[0]; a[15] = c[1];
  return a;
}

// 16x32 bf16 A-tile slice from a bf16 row (LDS or global), 16B vector loads.
static __device__ __forceinline__ v16bf load_a_bf16(const __bf16* row, int kbase,
                                                    int khalf) {
  v8bf lo = *(const v8bf*)(row + kbase + khalf * 16);
  v8bf hi = *(const v8bf*)(row + kbase + khalf * 16 + 8);
  v16bf a;
#pragma unroll
  for (int i = 0; i < 8; ++i) { a[i] = lo[i]; a[8 + i] = hi[i]; }
  return a;
}

// ---------------------------------------------------------------------------
// Kernel 0: transpose + convert weight fp32 [K][N] -> bf16 [N][K]
// (makes the WMMA B-operand a contiguous 32B per-lane load)
// ---------------------------------------------------------------------------
__global__ __launch_bounds__(256) void transpose_bf16_kernel(
    const float* __restrict__ src, __bf16* __restrict__ dst, int K, int N) {
  size_t i = (size_t)blockIdx.x * 256 + threadIdx.x;
  size_t total = (size_t)K * N;
  if (i < total) {
    int k = (int)(i / N);
    int n = (int)(i % N);
    dst[(size_t)n * K + k] = f2bf(src[i]);
  }
}

// ---------------------------------------------------------------------------
// Kernel 1: Q projection  q_bf16[16][512] = bf16(query @ Wq)
// ---------------------------------------------------------------------------
__global__ __launch_bounds__(128) void qproj_kernel(
    const float* __restrict__ query, const __bf16* __restrict__ Wqt,
    __bf16* __restrict__ qbf) {
  int wave = threadIdx.x >> 5, lane = threadIdx.x & 31;
  int nt = blockIdx.x * 4 + wave;          // 0..31
  int khalf = lane >> 4, m = lane & 15;
  const float* arow = query + (size_t)m * D_;
  const __bf16* brow = Wqt + (size_t)(nt * 16 + m) * D_;
  v8f acc = {};
  for (int kk = 0; kk < 16; ++kk) {
    v16bf a = load_a_f32(arow, kk * 32, khalf);
    v16bf b = *(const v16bf*)(brow + kk * 32 + khalf * 16);
    acc = wmma_bf16(a, b, acc);
  }
  int mrow = (lane >= 16) ? 8 : 0;
  int col = nt * 16 + m;
#pragma unroll
  for (int r = 0; r < 8; ++r) qbf[(size_t)(mrow + r) * D_ + col] = f2bf(acc[r]);
}

// ---------------------------------------------------------------------------
// Kernel 2: fused K/V projection + attention partials (flash-decoding chunks)
// grid = B*NKV*NCHUNK, block 256 (8 waves); wave -> 16 enc rows.
// Streams enc ONCE (the only large operand: 268 MB, memory-bound at 23.3 TB/s).
// Weights stay in global (256 KB total, L2-resident on a 192 MB L2).
// ---------------------------------------------------------------------------
__global__ __launch_bounds__(256) void attn_partial_kernel(
    const float* __restrict__ enc, const __bf16* __restrict__ Wkt,
    const __bf16* __restrict__ Wvt, const __bf16* __restrict__ qbf,
    float* __restrict__ pm, float* __restrict__ ps, float* __restrict__ pctx) {
  int bid = blockIdx.x;
  int chunk = bid % NCHUNK;
  int kv = (bid / NCHUNK) % NKV_;
  int b = bid / (NCHUNK * NKV_);

  int tid = threadIdx.x;
  int wave = tid >> 5, lane = tid & 31;
  int khalf = lane >> 4, m = lane & 15;

  __shared__ __align__(32) __bf16 Klds[CHUNK_T][HD_];     // [t][d]   16 KB
  __shared__ __align__(32) __bf16 VldsT[HD_][CHUNK_T];    // [d][t]   16 KB
  __shared__ __align__(32) __bf16 Pbf[16][CHUNK_T];       // A-op, rows 4..15 = 0
  __shared__ float Sc[NREP_][CHUNK_T];                    // raw scores, 2 KB

  // zero the padded probability A-operand rows once
  for (int i = tid; i < 16 * CHUNK_T; i += 256)
    ((__bf16*)Pbf)[i] = f2bf(0.0f);

  // ---- K/V tile projection: [16 rows, 64] = enc_tile[16,512] @ W^T -------
  int row = chunk * CHUNK_T + wave * 16 + m;              // global T row
  const float* encrow = enc + ((size_t)b * T_ + row) * D_;

  v8f kacc[4] = {};
  v8f vacc[4] = {};
  for (int kk = 0; kk < 16; ++kk) {
    __builtin_prefetch(encrow + kk * 32 + 128, 0, 1);     // global_prefetch_b8
    v16bf a = load_a_f32(encrow, kk * 32, khalf);
    int krow = kk * 32 + khalf * 16;
#pragma unroll
    for (int nt = 0; nt < 4; ++nt) {
      int ncol = kv * HD_ + nt * 16 + m;                  // column in KV_DIM
      v16bf bk = *(const v16bf*)(Wkt + (size_t)ncol * D_ + krow);
      kacc[nt] = wmma_bf16(a, bk, kacc[nt]);
      v16bf bv = *(const v16bf*)(Wvt + (size_t)ncol * D_ + krow);
      vacc[nt] = wmma_bf16(a, bv, vacc[nt]);
    }
  }

  // ---- spill K (row-major [t][d]) and V (transposed [d][t]) to LDS -------
  // C-layout: VGPR r holds row M=r (lanes<16) / M=r+8 (lanes>=16), col N=m.
  int mrow = (lane >= 16) ? 8 : 0;
#pragma unroll
  for (int nt = 0; nt < 4; ++nt) {
#pragma unroll
    for (int r = 0; r < 8; r += 2) {                      // packed cvt pairs
      v2bf kp = f2bf2(kacc[nt][r], kacc[nt][r + 1]);
      Klds[wave * 16 + mrow + r][nt * 16 + m]     = kp[0];
      Klds[wave * 16 + mrow + r + 1][nt * 16 + m] = kp[1];
    }
    v8bf vv;
#pragma unroll
    for (int r = 0; r < 8; r += 2) {
      v2bf vp = f2bf2(vacc[nt][r], vacc[nt][r + 1]);
      vv[r] = vp[0]; vv[r + 1] = vp[1];
    }
    // 8 contiguous t-values for fixed d -> single ds_store_b128
    *(v8bf*)(&VldsT[nt * 16 + m][wave * 16 + mrow]) = vv;
  }
  __syncthreads();

  // ---- scores: s[h,t] = q[h,:] . K[t,:]   (A = q heads, B = Klds^T) ------
  {
    bool valid = m < NREP_;
    int head = kv * NREP_ + (valid ? m : 0);
    const __bf16* qrow = qbf + (size_t)b * D_ + (size_t)head * HD_;
    v8f sacc = {};
#pragma unroll
    for (int ks = 0; ks < 2; ++ks) {
      v16bf a = load_a_bf16(qrow, ks * 32, khalf);
      if (!valid) {
#pragma unroll
        for (int i = 0; i < 16; ++i) a[i] = f2bf(0.0f);
      }
      v16bf bk = *(const v16bf*)(&Klds[wave * 16 + m][ks * 32 + khalf * 16]);
      sacc = wmma_bf16(a, bk, sacc);
    }
    if (lane < 16) {
#pragma unroll
      for (int h = 0; h < NREP_; ++h)
        Sc[h][wave * 16 + lane] = sacc[h] * SCALE_;
    }
  }
  __syncthreads();

  // ---- per-head chunk softmax partials (waves 0..3, one head each) -------
  if (wave < NREP_) {
    int h = wave;
    float v0 = Sc[h][lane], v1 = Sc[h][lane + 32];
    float v2 = Sc[h][lane + 64], v3 = Sc[h][lane + 96];
    float vmax = fmaxf(fmaxf(v0, v1), fmaxf(v2, v3));
#pragma unroll
    for (int off = 16; off > 0; off >>= 1)
      vmax = fmaxf(vmax, __shfl_xor(vmax, off, 32));
    float e0 = __expf(v0 - vmax), e1 = __expf(v1 - vmax);
    float e2 = __expf(v2 - vmax), e3 = __expf(v3 - vmax);
    float sum = e0 + e1 + e2 + e3;
#pragma unroll
    for (int off = 16; off > 0; off >>= 1) sum += __shfl_xor(sum, off, 32);
    v2bf p01 = f2bf2(e0, e1), p23 = f2bf2(e2, e3);
    Pbf[h][lane] = p01[0]; Pbf[h][lane + 32] = p01[1];
    Pbf[h][lane + 64] = p23[0]; Pbf[h][lane + 96] = p23[1];
    if (lane == 0) {
      int bh = b * NH_ + kv * NREP_ + h;
      pm[(size_t)bh * NCHUNK + chunk] = vmax;
      ps[(size_t)bh * NCHUNK + chunk] = sum;
    }
  }
  __syncthreads();

  // ---- partial context via WMMA: ctx[h][d] = sum_t P[h][t] * V[t][d] -----
  // A = Pbf [16(heads pad)][128], B = VldsT (k=t, n=d); wave w -> d-tile w.
  if (wave < NREP_) {
    const __bf16* prow = &Pbf[m][0];
    v8f cacc = {};
#pragma unroll
    for (int ks = 0; ks < 4; ++ks) {                      // K = 128 = 4 x 32
      v16bf a = load_a_bf16(prow, ks * 32, khalf);
      v16bf bv = *(const v16bf*)(&VldsT[wave * 16 + m][ks * 32 + khalf * 16]);
      cacc = wmma_bf16(a, bv, cacc);
    }
    if (lane < 16) {
      int d = wave * 16 + lane;
#pragma unroll
      for (int r = 0; r < NREP_; ++r) {                   // valid heads 0..3
        int bh = b * NH_ + kv * NREP_ + r;
        pctx[((size_t)bh * NCHUNK + chunk) * HD_ + d] = cacc[r];
      }
    }
  }
}

// ---------------------------------------------------------------------------
// Kernel 3: combine chunk partials (online softmax merge) -> hcat[:, 0:512]
// ---------------------------------------------------------------------------
__global__ __launch_bounds__(64) void reduce_kernel(
    const float* __restrict__ pm, const float* __restrict__ ps,
    const float* __restrict__ pctx, __bf16* __restrict__ hcat) {
  int bh = blockIdx.x;                 // 0..127
  int b = bh >> 3, h = bh & 7;
  int d = threadIdx.x;                 // 0..63
  float M = -3.0e38f, S = 0.0f, acc = 0.0f;
  for (int c = 0; c < NCHUNK; ++c) {
    float mc = pm[(size_t)bh * NCHUNK + c];
    float sc = ps[(size_t)bh * NCHUNK + c];
    float cx = pctx[((size_t)bh * NCHUNK + c) * HD_ + d];
    float nM = fmaxf(M, mc);
    float wOld = __expf(M - nM), wNew = __expf(mc - nM);
    S = S * wOld + sc * wNew;
    acc = acc * wOld + cx * wNew;
    M = nM;
  }
  hcat[(size_t)b * 1024 + h * HD_ + d] = f2bf(acc / S);
}

// Kernel 3b: hcat[:, 512:1024] = bf16(query)
__global__ __launch_bounds__(256) void qcopy_kernel(
    const float* __restrict__ query, __bf16* __restrict__ hcat) {
  int i = blockIdx.x * 256 + threadIdx.x;
  if (i < B_ * D_) {
    int b = i / D_, j = i % D_;
    hcat[(size_t)b * 1024 + 512 + j] = f2bf(query[i]);
  }
}

// ---------------------------------------------------------------------------
// Kernel 4: MLP1  act = silu(hcat @ W1)   [16,1024]@[1024,2048]
// A matrix (32 KB) is reused by all 4 waves -> async-stage into LDS once.
// ---------------------------------------------------------------------------
__global__ __launch_bounds__(128) void mlp1_kernel(
    const __bf16* __restrict__ hcat, const __bf16* __restrict__ W1t,
    __bf16* __restrict__ act) {
  __shared__ __align__(16) __bf16 Alds[16 * 1024];        // 32 KB
  int tid = threadIdx.x;
  for (int i = tid; i < 16 * 1024 / 8; i += 128)          // 16B per lane
    async_copy_b128(hcat + (size_t)i * 8, Alds + (size_t)i * 8);
  wait_async_zero();
  __syncthreads();

  int wave = tid >> 5, lane = tid & 31;
  int nt = blockIdx.x * 4 + wave;            // 0..127
  int khalf = lane >> 4, m = lane & 15;
  const __bf16* arow = Alds + (size_t)m * 1024;
  const __bf16* brow = W1t + (size_t)(nt * 16 + m) * 1024;
  v8f acc = {};
  for (int kk = 0; kk < 32; ++kk) {
    v16bf a = load_a_bf16(arow, kk * 32, khalf);
    v16bf b = *(const v16bf*)(brow + kk * 32 + khalf * 16);
    acc = wmma_bf16(a, b, acc);
  }
  int mrow = (lane >= 16) ? 8 : 0;
  int col = nt * 16 + m;
#pragma unroll
  for (int r = 0; r < 8; ++r) {
    float x = acc[r];
    float s = x / (1.0f + __expf(-x));       // SiLU on f32 accumulator
    act[(size_t)(mrow + r) * 2048 + col] = f2bf(s);
  }
}

// ---------------------------------------------------------------------------
// Kernel 5: MLP2  out = act @ W2   [16,2048]@[2048,512] -> fp32 d_out
// A matrix (64 KB) async-staged into LDS once per block.
// ---------------------------------------------------------------------------
__global__ __launch_bounds__(128) void mlp2_kernel(
    const __bf16* __restrict__ act, const __bf16* __restrict__ W2t,
    float* __restrict__ out) {
  __shared__ __align__(16) __bf16 Alds[16 * 2048];        // 64 KB
  int tid = threadIdx.x;
  for (int i = tid; i < 16 * 2048 / 8; i += 128)          // 16B per lane
    async_copy_b128(act + (size_t)i * 8, Alds + (size_t)i * 8);
  wait_async_zero();
  __syncthreads();

  int wave = tid >> 5, lane = tid & 31;
  int nt = blockIdx.x * 4 + wave;            // 0..31
  int khalf = lane >> 4, m = lane & 15;
  const __bf16* arow = Alds + (size_t)m * 2048;
  const __bf16* brow = W2t + (size_t)(nt * 16 + m) * 2048;
  v8f acc = {};
  for (int kk = 0; kk < 64; ++kk) {
    v16bf a = load_a_bf16(arow, kk * 32, khalf);
    v16bf b = *(const v16bf*)(brow + kk * 32 + khalf * 16);
    acc = wmma_bf16(a, b, acc);
  }
  int mrow = (lane >= 16) ? 8 : 0;
  int col = nt * 16 + m;
#pragma unroll
  for (int r = 0; r < 8; ++r) out[(size_t)(mrow + r) * D_ + col] = acc[r];
}

// ---------------------------------------------------------------------------
extern "C" void kernel_launch(void* const* d_in, const int* in_sizes, int n_in,
                              void* d_out, int out_size, void* d_ws, size_t ws_size,
                              hipStream_t stream) {
  (void)in_sizes; (void)n_in; (void)out_size; (void)ws_size;
  const float* query = (const float*)d_in[0];   // [16,512]
  const float* enc   = (const float*)d_in[1];   // [16,8192,512]
  const float* Wq    = (const float*)d_in[2];   // [512,512]
  const float* Wk    = (const float*)d_in[3];   // [512,128]
  const float* Wv    = (const float*)d_in[4];   // [512,128]
  const float* W1    = (const float*)d_in[5];   // [1024,2048]
  const float* W2    = (const float*)d_in[6];   // [2048,512]
  float* out = (float*)d_out;                   // [16,512] fp32

  char* ws = (char*)d_ws;
  size_t off = 0;
  auto alloc = [&](size_t bytes) -> char* {
    char* p = ws + off;
    off += (bytes + 255) & ~(size_t)255;
    return p;
  };
  __bf16* Wqt  = (__bf16*)alloc((size_t)512 * 512 * 2);
  __bf16* Wkt  = (__bf16*)alloc((size_t)128 * 512 * 2);
  __bf16* Wvt  = (__bf16*)alloc((size_t)128 * 512 * 2);
  __bf16* W1t  = (__bf16*)alloc((size_t)2048 * 1024 * 2);
  __bf16* W2t  = (__bf16*)alloc((size_t)512 * 2048 * 2);
  __bf16* qbf  = (__bf16*)alloc((size_t)B_ * D_ * 2);
  float*  pm   = (float*)alloc((size_t)B_ * NH_ * NCHUNK * 4);
  float*  ps   = (float*)alloc((size_t)B_ * NH_ * NCHUNK * 4);
  float*  pctx = (float*)alloc((size_t)B_ * NH_ * NCHUNK * HD_ * 4);
  __bf16* hcat = (__bf16*)alloc((size_t)B_ * 1024 * 2);
  __bf16* act  = (__bf16*)alloc((size_t)B_ * 2048 * 2);

  // weight prep: fp32 [K][N] -> bf16 [N][K]
  transpose_bf16_kernel<<<(512 * 512 + 255) / 256, 256, 0, stream>>>(Wq, Wqt, 512, 512);
  transpose_bf16_kernel<<<(512 * 128 + 255) / 256, 256, 0, stream>>>(Wk, Wkt, 512, 128);
  transpose_bf16_kernel<<<(512 * 128 + 255) / 256, 256, 0, stream>>>(Wv, Wvt, 512, 128);
  transpose_bf16_kernel<<<(1024 * 2048 + 255) / 256, 256, 0, stream>>>(W1, W1t, 1024, 2048);
  transpose_bf16_kernel<<<(2048 * 512 + 255) / 256, 256, 0, stream>>>(W2, W2t, 2048, 512);

  qproj_kernel<<<8, 128, 0, stream>>>(query, Wqt, qbf);
  attn_partial_kernel<<<B_ * NKV_ * NCHUNK, 256, 0, stream>>>(enc, Wkt, Wvt, qbf,
                                                              pm, ps, pctx);
  qcopy_kernel<<<(B_ * D_ + 255) / 256, 256, 0, stream>>>(query, hcat);
  reduce_kernel<<<B_ * NH_, 64, 0, stream>>>(pm, ps, pctx, hcat);
  mlp1_kernel<<<32, 128, 0, stream>>>(hcat, W1t, act);
  mlp2_kernel<<<8, 128, 0, stream>>>(act, W2t, out);
}
